// GnnModel_83365315215492
// MI455X (gfx1250) — matrix operand order repaired and checked by reading, not compiled, not run on hardware
//
#include <hip/hip_runtime.h>
#include <hip/hip_bf16.h>
#include <math.h>

// ---------------- types for WMMA ----------------
typedef __attribute__((ext_vector_type(16))) __bf16 v16bf;
typedef __attribute__((ext_vector_type(8)))  __bf16 v8bf;
typedef __attribute__((ext_vector_type(8)))  float  v8f;

// ---------------- problem constants (match reference) ----------------
constexpr int NFEAT = 500;
constexpr int NHID  = 64;
constexpr int NCLS  = 40;
constexpr int NCLSP = 48;   // NCLS padded to 3x16 tiles
constexpr int KS1   = 504;  // LDS column stride for W1 (16B aligned, bank-conflict-free)
constexpr int KS2   = 72;   // LDS column stride for W2 (16B aligned, bank-conflict-free)

// convert 8 f32 (two float4) into bf16 elements [off, off+8) of a v16bf
__device__ __forceinline__ void cvt8(v16bf& a, int off, float4 f0, float4 f1) {
    a[off + 0] = (__bf16)f0.x; a[off + 1] = (__bf16)f0.y;
    a[off + 2] = (__bf16)f0.z; a[off + 3] = (__bf16)f0.w;
    a[off + 4] = (__bf16)f1.x; a[off + 5] = (__bf16)f1.y;
    a[off + 6] = (__bf16)f1.z; a[off + 7] = (__bf16)f1.w;
}

// load bf16 of one column from LDS: K [kk, kk+8) and [kk+16, kk+24) (two ds_load_b128)
__device__ __forceinline__ v16bf loadB(const __bf16* lw, int stride, int col, int kk) {
    const v8bf* p = (const v8bf*)(lw + col * stride + kk);
    v8bf lo = p[0], hi = p[2];
    v16bf b;
    #pragma unroll
    for (int i = 0; i < 8; ++i) { b[i] = lo[i]; b[8 + i] = hi[i]; }
    return b;
}

// =====================================================================
// deg init (deg=1 for self loop) + zero loss accumulators
// =====================================================================
__global__ void init_kernel(float* __restrict__ deg, float* __restrict__ sums, int n) {
    int i = blockIdx.x * blockDim.x + threadIdx.x;
    if (i < n) deg[i] = 1.0f;
    if (i < 2) sums[i] = 0.0f;
}

__global__ void deg_kernel(const int* __restrict__ dst, float* __restrict__ deg, int E) {
    int e = blockIdx.x * blockDim.x + threadIdx.x;
    if (e < E) atomicAdd(&deg[dst[e]], 1.0f);
}

__global__ void dinv_kernel(float* __restrict__ deg, int n) {
    int i = blockIdx.x * blockDim.x + threadIdx.x;
    if (i < n) deg[i] = rsqrtf(deg[i]);
}

// =====================================================================
// GEMM1: H1 = X[n,500] @ W1[500,64]   (bf16 WMMA, f32 accumulate)
// one wave -> 16x64 tile; 8 waves/block; W1 column-major bf16 in LDS
// =====================================================================
__global__ __launch_bounds__(256)
void gemm1_kernel(const float* __restrict__ X, const float* __restrict__ W,
                  float* __restrict__ H, int n) {
    __shared__ __bf16 lw[NHID * KS1];   // 64*504*2 = 64512 B, [col][k]
    int tid = threadIdx.x;
    for (int idx = tid; idx < NHID * KS1; idx += 256) {
        int c = idx / KS1, k = idx % KS1;
        lw[idx] = (__bf16)((k < NFEAT) ? W[k * NHID + c] : 0.0f);
    }
    __syncthreads();

    int wave = tid >> 5, lane = tid & 31;
    int mtiles = (n + 15) >> 4;
    int rowTile = blockIdx.x * 8 + wave;
    if (rowTile >= mtiles) return;           // wave-uniform

    int half = lane >> 4, l16 = lane & 15;
    int row  = rowTile * 16 + l16;
    const float* xr = X + (size_t)((row < n) ? row : 0) * NFEAT;  // safe redirect
    int akbase = half * 8, bkbase = half * 8;

    v8f acc0 = {}, acc1 = {}, acc2 = {}, acc3 = {};

    // chunks 0..14: fully in-range, unconditional vector loads
    for (int kc = 0; kc < 15; ++kc) {
        int k0 = kc * 32;
        int base0 = k0 + akbase;
        const float4* p0 = (const float4*)(xr + base0);
        const float4* p1 = (const float4*)(xr + base0 + 16);
        float4 f0 = p0[0], f1 = p0[1];
        float4 f2 = p1[0], f3 = p1[1];
        v16bf a;
        cvt8(a, 0, f0, f1);
        cvt8(a, 8, f2, f3);

        int kkB = k0 + bkbase;
        v16bf b0 = loadB(lw, KS1, l16,      kkB);
        v16bf b1 = loadB(lw, KS1, l16 + 16, kkB);
        v16bf b2 = loadB(lw, KS1, l16 + 32, kkB);
        v16bf b3 = loadB(lw, KS1, l16 + 48, kkB);

        acc0 = __builtin_amdgcn_wmma_f32_16x16x32_bf16(false, a, false, b0, (short)0, acc0, false, false);
        acc1 = __builtin_amdgcn_wmma_f32_16x16x32_bf16(false, a, false, b1, (short)0, acc1, false, false);
        acc2 = __builtin_amdgcn_wmma_f32_16x16x32_bf16(false, a, false, b2, (short)0, acc2, false, false);
        acc3 = __builtin_amdgcn_wmma_f32_16x16x32_bf16(false, a, false, b3, (short)0, acc3, false, false);
    }

    // tail chunk kc=15 (K 480..511, real data ends at 499). A elements with
    // K>=500 are zeroed, which also cancels any stale/padded B values there.
    {
        int k0 = 480;
        int base0 = k0 + akbase;              // 480 or 488: runs 0..7 fully valid
        const float4* p0 = (const float4*)(xr + base0);
        float4 f0 = p0[0], f1 = p0[1];
        int base2 = base0 + 16;               // 496 (half0) or 504 (half1)
        float4 ft = make_float4(0.f, 0.f, 0.f, 0.f);
        if (base2 + 3 < NFEAT) ft = *(const float4*)(xr + base2);  // exactly 496..499
        float4 fz = make_float4(0.f, 0.f, 0.f, 0.f);
        v16bf a;
        cvt8(a, 0, f0, f1);
        cvt8(a, 8, ft, fz);

        int kkB = k0 + bkbase;
        v16bf b0 = loadB(lw, KS1, l16,      kkB);
        v16bf b1 = loadB(lw, KS1, l16 + 16, kkB);
        v16bf b2 = loadB(lw, KS1, l16 + 32, kkB);
        v16bf b3 = loadB(lw, KS1, l16 + 48, kkB);

        acc0 = __builtin_amdgcn_wmma_f32_16x16x32_bf16(false, a, false, b0, (short)0, acc0, false, false);
        acc1 = __builtin_amdgcn_wmma_f32_16x16x32_bf16(false, a, false, b1, (short)0, acc1, false, false);
        acc2 = __builtin_amdgcn_wmma_f32_16x16x32_bf16(false, a, false, b2, (short)0, acc2, false, false);
        acc3 = __builtin_amdgcn_wmma_f32_16x16x32_bf16(false, a, false, b3, (short)0, acc3, false, false);
    }

    #pragma unroll
    for (int r = 0; r < 8; ++r) {
        int m = rowTile * 16 + half * 8 + r;
        if (m < n) {
            float* hr = H + (size_t)m * NHID + l16;
            hr[0]  = acc0[r]; hr[16] = acc1[r];
            hr[32] = acc2[r]; hr[48] = acc3[r];
        }
    }
}

// =====================================================================
// self-loop seed: A1[i,c] = H1[i,c] * dinv[i]^2
// =====================================================================
__global__ void selfloop64_kernel(const float* __restrict__ H, const float* __restrict__ dinv,
                                  float* __restrict__ A, int n) {
    int idx = blockIdx.x * blockDim.x + threadIdx.x;
    if (idx < n * NHID) {
        int i = idx >> 6;
        float d = dinv[i];
        A[idx] = H[idx] * d * d;
    }
}

// edge scatter: A1[dst,c] += H1[src,c] * dinv[src]*dinv[dst]; 64 threads/edge
__global__ __launch_bounds__(256)
void agg64_kernel(const int* __restrict__ src, const int* __restrict__ dst,
                  const float* __restrict__ H, const float* __restrict__ dinv,
                  float* __restrict__ A, int E) {
    int t = blockIdx.x * 256 + threadIdx.x;
    int e = t >> 6;
    int c = t & 63;
    if (e < E) {
        int s = src[e], d = dst[e];
        float nm = dinv[s] * dinv[d];
        atomicAdd(&A[(size_t)d * NHID + c], H[(size_t)s * NHID + c] * nm);
    }
}

// =====================================================================
// GEMM2: H2 = relu(A1 + b1) @ W2[64,40]   (bf16 WMMA), N padded to 48
// =====================================================================
__global__ __launch_bounds__(256)
void gemm2_kernel(const float* __restrict__ A1, const float* __restrict__ b1,
                  const float* __restrict__ W2, float* __restrict__ H2, int n) {
    __shared__ __bf16 lw[NCLSP * KS2];   // 48*72*2 = 6912 B, [col][k]
    __shared__ float  lb[NHID];
    int tid = threadIdx.x;
    for (int idx = tid; idx < NCLSP * KS2; idx += 256) {
        int c = idx / KS2, k = idx % KS2;
        lw[idx] = (__bf16)((c < NCLS && k < NHID) ? W2[k * NCLS + c] : 0.0f);
    }
    if (tid < NHID) lb[tid] = b1[tid];
    __syncthreads();

    int wave = tid >> 5, lane = tid & 31;
    int mtiles = (n + 15) >> 4;
    int rowTile = blockIdx.x * 8 + wave;
    if (rowTile >= mtiles) return;

    int half = lane >> 4, l16 = lane & 15;
    int row  = rowTile * 16 + l16;
    const float* ar = A1 + (size_t)((row < n) ? row : 0) * NHID;
    int akbase = half * 8, bkbase = half * 8;

    v8f acc0 = {}, acc1 = {}, acc2 = {};

    #pragma unroll
    for (int kc = 0; kc < 2; ++kc) {        // K=64 -> 2 chunks of 32, no tail
        int k0 = kc * 32;
        int base0 = k0 + akbase;
        const float4* p0 = (const float4*)(ar + base0);
        const float4* p1 = (const float4*)(ar + base0 + 16);
        float4 f0 = p0[0], f1 = p0[1];
        float4 f2 = p1[0], f3 = p1[1];
        // relu(x + b1) on registers
        f0.x = fmaxf(f0.x + lb[base0 + 0], 0.f);  f0.y = fmaxf(f0.y + lb[base0 + 1], 0.f);
        f0.z = fmaxf(f0.z + lb[base0 + 2], 0.f);  f0.w = fmaxf(f0.w + lb[base0 + 3], 0.f);
        f1.x = fmaxf(f1.x + lb[base0 + 4], 0.f);  f1.y = fmaxf(f1.y + lb[base0 + 5], 0.f);
        f1.z = fmaxf(f1.z + lb[base0 + 6], 0.f);  f1.w = fmaxf(f1.w + lb[base0 + 7], 0.f);
        f2.x = fmaxf(f2.x + lb[base0 + 16], 0.f); f2.y = fmaxf(f2.y + lb[base0 + 17], 0.f);
        f2.z = fmaxf(f2.z + lb[base0 + 18], 0.f); f2.w = fmaxf(f2.w + lb[base0 + 19], 0.f);
        f3.x = fmaxf(f3.x + lb[base0 + 20], 0.f); f3.y = fmaxf(f3.y + lb[base0 + 21], 0.f);
        f3.z = fmaxf(f3.z + lb[base0 + 22], 0.f); f3.w = fmaxf(f3.w + lb[base0 + 23], 0.f);
        v16bf a;
        cvt8(a, 0, f0, f1);
        cvt8(a, 8, f2, f3);

        int kkB = k0 + bkbase;
        v16bf b0 = loadB(lw, KS2, l16,      kkB);
        v16bf b1v = loadB(lw, KS2, l16 + 16, kkB);
        v16bf b2 = loadB(lw, KS2, l16 + 32, kkB);

        acc0 = __builtin_amdgcn_wmma_f32_16x16x32_bf16(false, a, false, b0,  (short)0, acc0, false, false);
        acc1 = __builtin_amdgcn_wmma_f32_16x16x32_bf16(false, a, false, b1v, (short)0, acc1, false, false);
        acc2 = __builtin_amdgcn_wmma_f32_16x16x32_bf16(false, a, false, b2,  (short)0, acc2, false, false);
    }

    #pragma unroll
    for (int r = 0; r < 8; ++r) {
        int m = rowTile * 16 + half * 8 + r;
        if (m < n) {
            float* hr = H2 + (size_t)m * NCLS;
            hr[l16]      = acc0[r];
            hr[l16 + 16] = acc1[r];
            if (l16 + 32 < NCLS) hr[l16 + 32] = acc2[r];
        }
    }
}

// self-loop seed, 40-wide
__global__ void selfloop40_kernel(const float* __restrict__ H, const float* __restrict__ dinv,
                                  float* __restrict__ A, int n) {
    int idx = blockIdx.x * blockDim.x + threadIdx.x;
    if (idx < n * NCLS) {
        int i = idx / NCLS;
        float d = dinv[i];
        A[idx] = H[idx] * d * d;
    }
}

// edge scatter, 40 active of 64 threads/edge
__global__ __launch_bounds__(256)
void agg40_kernel(const int* __restrict__ src, const int* __restrict__ dst,
                  const float* __restrict__ H, const float* __restrict__ dinv,
                  float* __restrict__ A, int E) {
    int t = blockIdx.x * 256 + threadIdx.x;
    int e = t >> 6;
    int c = t & 63;
    if (e < E && c < NCLS) {
        int s = src[e], d = dst[e];
        float nm = dinv[s] * dinv[d];
        atomicAdd(&A[(size_t)d * NCLS + c], H[(size_t)s * NCLS + c] * nm);
    }
}

// =====================================================================
// masked cross-entropy: sums[0] += nll*mask, sums[1] += mask
// =====================================================================
__global__ __launch_bounds__(256)
void loss_kernel(const float* __restrict__ A2, const float* __restrict__ b2,
                 const int* __restrict__ y, const int* __restrict__ mask,
                 float* __restrict__ sums, int n) {
    __shared__ float lb2[NCLS];
    if (threadIdx.x < NCLS) lb2[threadIdx.x] = b2[threadIdx.x];
    __syncthreads();
    int i = blockIdx.x * 256 + threadIdx.x;
    if (i >= n) return;
    const float* row = A2 + (size_t)i * NCLS;
    float mx = -1e30f;
    float l[NCLS];
    #pragma unroll 8
    for (int c = 0; c < NCLS; ++c) { l[c] = row[c] + lb2[c]; mx = fmaxf(mx, l[c]); }
    float s = 0.0f;
    #pragma unroll 8
    for (int c = 0; c < NCLS; ++c) s += __expf(l[c] - mx);
    float lse = __logf(s) + mx;
    int yy = y[i];
    float nll = lse - l[yy];
    if (mask[i] != 0) {
        atomicAdd(&sums[0], nll);
        atomicAdd(&sums[1], 1.0f);
    }
}

__global__ void finalize_kernel(const float* __restrict__ sums, float* __restrict__ out) {
    out[0] = sums[0] / sums[1];
}

// =====================================================================
// launch
// =====================================================================
extern "C" void kernel_launch(void* const* d_in, const int* in_sizes, int n_in,
                              void* d_out, int out_size, void* d_ws, size_t ws_size,
                              hipStream_t stream) {
    const float* x    = (const float*)d_in[0];
    const float* W1   = (const float*)d_in[1];
    const float* b1   = (const float*)d_in[2];
    const float* W2   = (const float*)d_in[3];
    const float* b2   = (const float*)d_in[4];
    const int*   ei   = (const int*)d_in[5];   // [2, E] flat
    const int*   y    = (const int*)d_in[6];
    const int*   mask = (const int*)d_in[7];
    float* out = (float*)d_out;

    int n = in_sizes[6];
    int E = in_sizes[5] / 2;
    const int* src = ei;
    const int* dst = ei + E;

    float* ws = (float*)d_ws;
    size_t off = 0;
    auto take = [&](size_t cnt) { float* p = ws + off; off += (cnt + 255) & ~(size_t)255; return p; };
    float* deg  = take((size_t)n);            // -> dinv in place
    float* H1   = take((size_t)n * NHID);
    float* A1   = take((size_t)n * NHID);
    float* H2   = take((size_t)n * NCLS);
    float* A2   = take((size_t)n * NCLS);
    float* sums = take(2);
    (void)ws_size;

    int mtiles = (n + 15) / 16;
    int gemmBlocks = (mtiles + 7) / 8;

    init_kernel<<<(n + 255) / 256, 256, 0, stream>>>(deg, sums, n);
    deg_kernel<<<(E + 255) / 256, 256, 0, stream>>>(dst, deg, E);
    dinv_kernel<<<(n + 255) / 256, 256, 0, stream>>>(deg, n);

    gemm1_kernel<<<gemmBlocks, 256, 0, stream>>>(x, W1, H1, n);
    selfloop64_kernel<<<((size_t)n * NHID + 255) / 256, 256, 0, stream>>>(H1, deg, A1, n);
    agg64_kernel<<<((size_t)E * 64 + 255) / 256, 256, 0, stream>>>(src, dst, H1, deg, A1, E);

    gemm2_kernel<<<gemmBlocks, 256, 0, stream>>>(A1, b1, W2, H2, n);
    selfloop40_kernel<<<((size_t)n * NCLS + 255) / 256, 256, 0, stream>>>(H2, deg, A2, n);
    agg40_kernel<<<((size_t)E * 64 + 255) / 256, 256, 0, stream>>>(src, dst, H2, deg, A2, E);

    loss_kernel<<<(n + 255) / 256, 256, 0, stream>>>(A2, b2, y, mask, sums, n);
    finalize_kernel<<<1, 1, 0, stream>>>(sums, out);
}
